// FluxIPAttnProcessor_44117904064817
// MI455X (gfx1250) — matrix operand order repaired and checked by reading, not compiled, not run on hardware
//
#include <hip/hip_runtime.h>

#define H_HEADS 24
#define DHEAD   128
#define DMODEL  3072
#define S_IMG   2048
#define L_TXT   512
#define S_ALL   2560
#define N_IP    1024

// Set to 0 if the assembler rejects the async mnemonics.
#define USE_ASYNC_COPY 1

typedef __attribute__((ext_vector_type(16))) __bf16 bf16x16;
typedef __attribute__((ext_vector_type(8)))  float  v8f;

union ABf16 { unsigned u[8]; uint4 q[2]; bf16x16 v; };

__device__ __forceinline__ v8f v8f_zero() {
  v8f z = {0.f, 0.f, 0.f, 0.f, 0.f, 0.f, 0.f, 0.f};
  return z;
}

__device__ __forceinline__ v8f wmma_bf16(const ABf16& a, const ABf16& b, v8f c) {
  return __builtin_amdgcn_wmma_f32_16x16x32_bf16(false, a.v, false, b.v,
                                                 (short)0, c, false, false);
}

// 16-byte global -> LDS transfer.  CDNA5 async path (ASYNCcnt-tracked DMA into
// LDS, CDNA5 ISA) with a plain-copy fallback.
__device__ __forceinline__ void copy16_g2l(const void* g, void* l) {
#if USE_ASYNC_COPY
  const unsigned lofs = (unsigned)(size_t)l;            // LDS byte offset
  const unsigned long long ga = (unsigned long long)(size_t)g;
  asm volatile("global_load_async_to_lds_b128 %0, %1, off"
               :: "v"(lofs), "v"(ga) : "memory");
#else
  *(uint4*)l = *(const uint4*)g;
#endif
}
__device__ __forceinline__ void copy_g2l_wait() {
#if USE_ASYNC_COPY
  asm volatile("s_wait_asynccnt 0x0" ::: "memory");
#endif
}

// ---------------------------------------------------------------------------
// Weight transpose + bf16 convert: W f32 [K][N] -> WT bf16 [N][K].
// Makes the GEMM B-operand K-contiguous so B fragments are 2x ds_load_b128.
// ---------------------------------------------------------------------------
__global__ void transpose_w_kernel(const float* __restrict__ W,
                                   __bf16* __restrict__ WT,
                                   int K, int N) {
  __shared__ __bf16 t[64][72];                 // padded vs bank conflicts
  const int k0 = blockIdx.y * 64;
  const int n0 = blockIdx.x * 64;
  const int r  = threadIdx.x >> 2;             // 0..63
  const int c0 = (threadIdx.x & 3) * 16;       // 0,16,32,48
  const float* src = W + (size_t)(k0 + r) * N + n0 + c0;
#pragma unroll
  for (int i = 0; i < 16; ++i) t[c0 + i][r] = (__bf16)src[i];
  __syncthreads();
  __bf16* dst = WT + (size_t)(n0 + r) * K + k0 + c0;
#pragma unroll
  for (int i = 0; i < 16; ++i) dst[i] = t[r][c0 + i];
}

// ---------------------------------------------------------------------------
// GEMM: C[M,N] = A[M,K] @ W[K,N] + bias, with W given as bf16 WT[N][K].
// Block 256 (8 waves), tile 128x128, K-step 32.  All LDS fragment loads are
// 2x ds_load_b128; all LDS stores are contiguous.
// ---------------------------------------------------------------------------
__global__ void gemm_bias_kernel(const float* __restrict__ A,
                                 const __bf16* __restrict__ WT,
                                 const float* __restrict__ bias,
                                 float* __restrict__ C,
                                 int M, int N, int K) {
  extern __shared__ char smraw[];
  __bf16* As = (__bf16*)smraw;        // [128 m][32 k]
  __bf16* Bs = As + 128 * 32;         // [128 n][32 k]

  const int tid  = threadIdx.x;
  const int wave = tid >> 5;
  const int lane = tid & 31;
  const int ln   = lane & 15;
  const int lh   = lane >> 4;
  const int tileM = blockIdx.y * 128;
  const int tileN = blockIdx.x * 128;
  const int wm = (wave >> 1) * 32;
  const int wn = (wave & 1) * 64;

  v8f acc[2][4];
#pragma unroll
  for (int i = 0; i < 2; ++i)
#pragma unroll
    for (int j = 0; j < 4; ++j) acc[i][j] = v8f_zero();

  const int row = tid >> 1;                 // 0..127 (A row / B n-row)
  const int col0 = (tid & 1) * 16;          // 0,16

  for (int k0 = 0; k0 < K; k0 += 32) {
    const float* asrc = A + (size_t)(tileM + row) * K + k0 + col0;
    __bf16* adst = As + row * 32 + col0;
#pragma unroll
    for (int i = 0; i < 16; ++i) adst[i] = (__bf16)asrc[i];   // v_cvt_pk_bf16_f32

    const __bf16* bsrc = WT + (size_t)(tileN + row) * K + k0 + col0;
    __bf16* bdst = Bs + row * 32 + col0;
    *(uint4*)bdst       = *(const uint4*)bsrc;
    *(uint4*)(bdst + 8) = *(const uint4*)(bsrc + 8);

    if (k0 + 32 < K) {                       // global_prefetch_b8
      __builtin_prefetch(asrc + 32, 0, 0);
      __builtin_prefetch(bsrc + 32, 0, 0);
    }
    __syncthreads();

    ABf16 af[2];
#pragma unroll
    for (int i = 0; i < 2; ++i) {
      const char* ap = (const char*)(As + (wm + i * 16 + ln) * 32);
      af[i].q[0] = *(const uint4*)(ap + lh * 16);        // K 0..15 half
      af[i].q[1] = *(const uint4*)(ap + 32 + lh * 16);   // K 16..31 half
    }
#pragma unroll
    for (int j = 0; j < 4; ++j) {
      ABf16 bfv;
      const char* bp = (const char*)(Bs + (wn + j * 16 + ln) * 32);
      bfv.q[0] = *(const uint4*)(bp + lh * 16);
      bfv.q[1] = *(const uint4*)(bp + 32 + lh * 16);
#pragma unroll
      for (int i = 0; i < 2; ++i) acc[i][j] = wmma_bf16(af[i], bfv, acc[i][j]);
    }
    __syncthreads();
  }

#pragma unroll
  for (int i = 0; i < 2; ++i)
#pragma unroll
    for (int j = 0; j < 4; ++j) {
      const int n = tileN + wn + j * 16 + ln;
      const float bv = bias ? bias[n] : 0.0f;
#pragma unroll
      for (int r = 0; r < 8; ++r) {
        const int m = tileM + wm + i * 16 + lh * 8 + r;
        C[(size_t)m * N + n] = acc[i][j][r] + bv;
      }
    }
}

// ---------------------------------------------------------------------------
// Per-head RMSNorm (+ optional RoPE) + reshape [S,3072] -> bf16 [H][S_dst][128]
// ---------------------------------------------------------------------------
__global__ void pack_head_kernel(const float* __restrict__ src,
                                 const float* __restrict__ rms_w,
                                 const float* __restrict__ cosb,
                                 const float* __restrict__ sinb,
                                 int rope_off,
                                 __bf16* __restrict__ dst,
                                 int S_dst, int row0) {
  const int s = blockIdx.x;
  const int h = blockIdx.y;
  const int d = threadIdx.x;            // 0..127
  const float* x = src + (size_t)s * DMODEL + h * DHEAD;

  float xd = x[d];
  float inv = 1.0f;
  __shared__ float red[4];
  if (rms_w) {
    float ss = xd * xd;
#pragma unroll
    for (int m = 1; m < 32; m <<= 1) ss += __shfl_xor(ss, m, 32);
    if ((d & 31) == 0) red[d >> 5] = ss;
    __syncthreads();
    inv = rsqrtf((red[0] + red[1] + red[2] + red[3]) * (1.0f / 128.0f) + 1e-6f);
  }
  const float wv = rms_w ? rms_w[d] : 1.0f;
  float v = xd * inv * wv;
  if (cosb) {
    const float wp = rms_w ? rms_w[d ^ 1] : 1.0f;
    const float partner = x[d ^ 1] * inv * wp;
    const float c  = cosb[(size_t)(rope_off + s) * DHEAD + d];
    const float sn = sinb[(size_t)(rope_off + s) * DHEAD + d];
    const float xrot = (d & 1) ? partner : -partner;
    v = v * c + xrot * sn;
  }
  dst[((size_t)h * S_dst + row0 + s) * DHEAD + d] = (__bf16)v;
}

// ---------------------------------------------------------------------------
// V pack, head-TRANSPOSED: f32 [S,3072] -> bf16 [H][128 dim][S_dst key]
// so the P@V B-operand is key-contiguous (2x ds_load_b128 fragments).
// Grid (Srows/64, H), block 256.
// ---------------------------------------------------------------------------
__global__ void pack_vT_kernel(const float* __restrict__ src,
                               __bf16* __restrict__ dst,
                               int S_dst, int row0) {
  __shared__ __bf16 t[128][72];               // [dim][64 s] padded
  const int s0 = blockIdx.x * 64;
  const int h  = blockIdx.y;
  {
    const int sl = threadIdx.x >> 2;          // 0..63
    const int d0 = (threadIdx.x & 3) * 32;    // 0,32,64,96
    const float* p = src + (size_t)(s0 + sl) * DMODEL + h * DHEAD + d0;
#pragma unroll
    for (int i = 0; i < 32; ++i) t[d0 + i][sl] = (__bf16)p[i];
  }
  __syncthreads();
  {
    const int d    = threadIdx.x >> 1;        // 0..127
    const int sseg = (threadIdx.x & 1) * 32;  // 0,32
    __bf16* q = dst + ((size_t)h * DHEAD + d) * S_dst + row0 + s0 + sseg;
#pragma unroll
    for (int i = 0; i < 32; ++i) q[i] = t[d][sseg + i];
  }
}

// ---------------------------------------------------------------------------
// Flash attention.  Q,K bf16 [H][S][128]; V bf16 head-transposed [H][128][Skv].
// Block 256 (8 waves) = (head, 128 q rows); wave = 16 rows.  K/V tiles filled
// with global_load_async_to_lds_b128; all WMMA fragments are 2x ds_load_b128.
// LDS: K 32KB + V 32KB + P staging 32KB = 96KB.
// ---------------------------------------------------------------------------
__global__ void flash_attn_kernel(const __bf16* __restrict__ Qh,
                                  const __bf16* __restrict__ Kh,
                                  const __bf16* __restrict__ Vt,
                                  float* __restrict__ Oh,
                                  int Sq, int Skv, float scale) {
  extern __shared__ char smraw[];
  __bf16* Ks = (__bf16*)smraw;             // [128 key][128 dim]
  __bf16* Vs = Ks + 128 * 128;             // [128 dim][128 key]
  __bf16* Ps = Vs + 128 * 128;             // [8 waves][16 m][128 key]

  const int tid  = threadIdx.x;
  const int wave = tid >> 5;
  const int lane = tid & 31;
  const int ln   = lane & 15;
  const int lh   = lane >> 4;
  const int h    = blockIdx.y;
  const int q0   = blockIdx.x * 128 + wave * 16;

  const __bf16* Qb  = Qh + (size_t)h * Sq * DHEAD;
  const __bf16* Kb  = Kh + (size_t)h * Skv * DHEAD;
  const __bf16* Vth = Vt + (size_t)h * DHEAD * Skv;

  // Persistent Q fragments (A-operand, 4 K-steps of 32 along head dim)
  ABf16 qf[4];
  {
    const char* qp = (const char*)(Qb + (size_t)(q0 + ln) * DHEAD);
#pragma unroll
    for (int kk = 0; kk < 4; ++kk) {
      qf[kk].q[0] = *(const uint4*)(qp + kk * 64 + lh * 16);
      qf[kk].q[1] = *(const uint4*)(qp + kk * 64 + 32 + lh * 16);
    }
  }

  float m_run[8], l_run[8];
#pragma unroll
  for (int r = 0; r < 8; ++r) { m_run[r] = -3.0e38f; l_run[r] = 0.0f; }
  v8f o[8];
#pragma unroll
  for (int t = 0; t < 8; ++t) o[t] = v8f_zero();

  __bf16* pw = Ps + wave * 16 * 128;
  const int trow = tid >> 1;                 // 0..127
  const int tseg = (tid & 1) * 64;           // 0,64

  for (int j0 = 0; j0 < Skv; j0 += 128) {
    { // async fill: K tile rows = keys; V tile rows = dims (already transposed)
      const __bf16* ksrc = Kb + (size_t)(j0 + trow) * DHEAD + tseg;
      __bf16*       kdst = Ks + trow * 128 + tseg;
      const __bf16* vsrc = Vth + (size_t)trow * Skv + j0 + tseg;
      __bf16*       vdst = Vs + trow * 128 + tseg;
#pragma unroll
      for (int i = 0; i < 8; ++i) {
        copy16_g2l(ksrc + i * 8, kdst + i * 8);
        copy16_g2l(vsrc + i * 8, vdst + i * 8);
      }
      copy_g2l_wait();
    }
    __syncthreads();

    // S = Q @ K^T  (N = key, GEMM-K = head dim)
    v8f sacc[8];
#pragma unroll
    for (int t = 0; t < 8; ++t) {
      sacc[t] = v8f_zero();
      const char* kp = (const char*)(Ks + (t * 16 + ln) * 128);
#pragma unroll
      for (int kk = 0; kk < 4; ++kk) {
        ABf16 bfv;
        bfv.q[0] = *(const uint4*)(kp + kk * 64 + lh * 16);
        bfv.q[1] = *(const uint4*)(kp + kk * 64 + 32 + lh * 16);
        sacc[t] = wmma_bf16(qf[kk], bfv, sacc[t]);
      }
#pragma unroll
      for (int r = 0; r < 8; ++r) sacc[t][r] *= scale;
    }

    // Online softmax (a D-tile row spans the 16 lanes of a half-wave)
#pragma unroll
    for (int r = 0; r < 8; ++r) {
      float mx = sacc[0][r];
#pragma unroll
      for (int t = 1; t < 8; ++t) mx = fmaxf(mx, sacc[t][r]);
#pragma unroll
      for (int msk = 1; msk < 16; msk <<= 1) mx = fmaxf(mx, __shfl_xor(mx, msk, 32));
      const float mnew  = fmaxf(m_run[r], mx);
      const float alpha = __expf(m_run[r] - mnew);
      float rs = 0.0f;
#pragma unroll
      for (int t = 0; t < 8; ++t) {
        const float p = __expf(sacc[t][r] - mnew);
        sacc[t][r] = p;
        rs += p;
      }
#pragma unroll
      for (int msk = 1; msk < 16; msk <<= 1) rs += __shfl_xor(rs, msk, 32);
      l_run[r] = l_run[r] * alpha + rs;
      m_run[r] = mnew;
#pragma unroll
      for (int t = 0; t < 8; ++t) o[t][r] *= alpha;
    }

    // Stage P (bf16) per-wave so it can be re-read in A-fragment layout
#pragma unroll
    for (int t = 0; t < 8; ++t)
#pragma unroll
      for (int r = 0; r < 8; ++r)
        pw[(lh * 8 + r) * 128 + t * 16 + ln] = (__bf16)sacc[t][r];
    __syncthreads();

    ABf16 af[4];
    {
      const char* pp = (const char*)(pw + ln * 128);
#pragma unroll
      for (int kk = 0; kk < 4; ++kk) {
        af[kk].q[0] = *(const uint4*)(pp + kk * 64 + lh * 16);
        af[kk].q[1] = *(const uint4*)(pp + kk * 64 + 32 + lh * 16);
      }
    }

    // O += P @ V  (N = head dim, GEMM-K = key; Vs is [dim][key])
#pragma unroll
    for (int td = 0; td < 8; ++td) {
      const char* vp = (const char*)(Vs + (td * 16 + ln) * 128);
#pragma unroll
      for (int kk = 0; kk < 4; ++kk) {
        ABf16 bfv;
        bfv.q[0] = *(const uint4*)(vp + kk * 64 + lh * 16);
        bfv.q[1] = *(const uint4*)(vp + kk * 64 + 32 + lh * 16);
        o[td] = wmma_bf16(af[kk], bfv, o[td]);
      }
    }
    __syncthreads();
  }

  float* Ob = Oh + (size_t)h * Sq * DHEAD;
#pragma unroll
  for (int td = 0; td < 8; ++td)
#pragma unroll
    for (int r = 0; r < 8; ++r) {
      const int m = q0 + lh * 8 + r;
      Ob[(size_t)m * DHEAD + td * 16 + ln] = o[td][r] / l_run[r];
    }
}

// ---------------------------------------------------------------------------
// Merge head-major outputs back to [S,3072]; add IP residual on img rows.
// ---------------------------------------------------------------------------
__global__ void merge_add_kernel(const float* __restrict__ O,
                                 const float* __restrict__ Oip,
                                 float* __restrict__ img,
                                 float* __restrict__ txt) {
  const size_t i = (size_t)blockIdx.x * 256 + threadIdx.x;
  const int s = (int)(i / DMODEL);
  const int c = (int)(i % DMODEL);
  const int h = c >> 7;
  const int d = c & 127;
  const float val = O[((size_t)h * S_ALL + s) * DHEAD + d];
  if (s < L_TXT) {
    txt[(size_t)s * DMODEL + c] = val;
  } else {
    const int si = s - L_TXT;
    img[(size_t)si * DMODEL + c] = val + Oip[((size_t)h * S_IMG + si) * DHEAD + d];
  }
}

// ---------------------------------------------------------------------------
extern "C" void kernel_launch(void* const* d_in, const int* in_sizes, int n_in,
                              void* d_out, int out_size, void* d_ws, size_t ws_size,
                              hipStream_t stream) {
  (void)in_sizes; (void)n_in; (void)out_size; (void)ws_size;

  const float* hidden = (const float*)d_in[0];
  const float* enc    = (const float*)d_in[1];
  const float* iph    = (const float*)d_in[2];
  const float* rcos   = (const float*)d_in[3];
  const float* rsin   = (const float*)d_in[4];
  const float* Wq  = (const float*)d_in[5];   const float* bq  = (const float*)d_in[6];
  const float* Wk  = (const float*)d_in[7];   const float* bk  = (const float*)d_in[8];
  const float* Wv  = (const float*)d_in[9];   const float* bv  = (const float*)d_in[10];
  const float* nq  = (const float*)d_in[11];  const float* nk  = (const float*)d_in[12];
  const float* aWq = (const float*)d_in[13];  const float* abq = (const float*)d_in[14];
  const float* aWk = (const float*)d_in[15];  const float* abk = (const float*)d_in[16];
  const float* aWv = (const float*)d_in[17];  const float* abv = (const float*)d_in[18];
  const float* naq = (const float*)d_in[19];  const float* nak = (const float*)d_in[20];
  const float* Wkip = (const float*)d_in[21]; const float* bkip = (const float*)d_in[22];
  const float* Wvip = (const float*)d_in[23]; const float* bvip = (const float*)d_in[24];
  const float* nipq = (const float*)d_in[25]; const float* nipk = (const float*)d_in[26];
  const float* Wo  = (const float*)d_in[27];  const float* bo  = (const float*)d_in[28];
  const float* Wao = (const float*)d_in[29];  const float* bao = (const float*)d_in[30];

  char* ws = (char*)d_ws;
  size_t off = 0;
  auto alloc = [&](size_t bytes) -> void* {
    void* p = (void*)(ws + off);
    off += (bytes + 255) & ~((size_t)255);
    return p;
  };

  float* Q0 = (float*)alloc((size_t)S_IMG * DMODEL * 4);
  float* K0 = (float*)alloc((size_t)S_IMG * DMODEL * 4);
  float* V0 = (float*)alloc((size_t)S_IMG * DMODEL * 4);
  float* EQ = (float*)alloc((size_t)L_TXT * DMODEL * 4);
  float* EK = (float*)alloc((size_t)L_TXT * DMODEL * 4);
  float* EV = (float*)alloc((size_t)L_TXT * DMODEL * 4);
  float* PK = (float*)alloc((size_t)N_IP * DMODEL * 4);
  float* PV = (float*)alloc((size_t)N_IP * DMODEL * 4);
  __bf16* WTs = (__bf16*)alloc((size_t)DMODEL * DMODEL * 2);   // reused scratch
  __bf16* Qb  = (__bf16*)alloc((size_t)H_HEADS * S_ALL * DHEAD * 2);
  __bf16* Kb  = (__bf16*)alloc((size_t)H_HEADS * S_ALL * DHEAD * 2);
  __bf16* VbT = (__bf16*)alloc((size_t)H_HEADS * DHEAD * S_ALL * 2);
  __bf16* IPQ = (__bf16*)alloc((size_t)H_HEADS * S_IMG * DHEAD * 2);
  __bf16* IPK = (__bf16*)alloc((size_t)H_HEADS * N_IP * DHEAD * 2);
  __bf16* IPVT= (__bf16*)alloc((size_t)H_HEADS * DHEAD * N_IP * 2);
  float* O   = (float*)alloc((size_t)H_HEADS * S_ALL * DHEAD * 4);
  float* Oip = (float*)alloc((size_t)H_HEADS * S_IMG * DHEAD * 4);
  float* Timg = Q0;   // Q0/K0 dead after pack stage
  float* Ttxt = K0;

  float* img_out = (float*)d_out;
  float* enc_out = (float*)d_out + (size_t)S_IMG * DMODEL;

  const dim3 b256(256);
  const size_t gemm_lds = (size_t)(128 * 32 * 2) * 2;   // As + Bs, 16 KB

  auto tgemm = [&](const float* A, const float* W, const float* b, float* C,
                   int M, int N, int K) {
    transpose_w_kernel<<<dim3(N / 64, K / 64), b256, 0, stream>>>(W, WTs, K, N);
    gemm_bias_kernel<<<dim3(N / 128, M / 128), b256, gemm_lds, stream>>>(A, WTs, b, C, M, N, K);
  };

  // Projections
  tgemm(hidden, Wq, bq, Q0, S_IMG, DMODEL, DMODEL);
  tgemm(hidden, Wk, bk, K0, S_IMG, DMODEL, DMODEL);
  tgemm(hidden, Wv, bv, V0, S_IMG, DMODEL, DMODEL);
  tgemm(enc, aWq, abq, EQ, L_TXT, DMODEL, DMODEL);
  tgemm(enc, aWk, abk, EK, L_TXT, DMODEL, DMODEL);
  tgemm(enc, aWv, abv, EV, L_TXT, DMODEL, DMODEL);
  tgemm(iph, Wkip, bkip, PK, N_IP, DMODEL, 2048);
  tgemm(iph, Wvip, bvip, PV, N_IP, DMODEL, 2048);

  // RMSNorm / RoPE / packing (Q,K row-major bf16; V head-transposed bf16)
  pack_head_kernel<<<dim3(L_TXT, H_HEADS), dim3(128), 0, stream>>>(EQ, naq, rcos, rsin, 0,     Qb, S_ALL, 0);
  pack_head_kernel<<<dim3(S_IMG, H_HEADS), dim3(128), 0, stream>>>(Q0, nq,  rcos, rsin, L_TXT, Qb, S_ALL, L_TXT);
  pack_head_kernel<<<dim3(L_TXT, H_HEADS), dim3(128), 0, stream>>>(EK, nak, rcos, rsin, 0,     Kb, S_ALL, 0);
  pack_head_kernel<<<dim3(S_IMG, H_HEADS), dim3(128), 0, stream>>>(K0, nk,  rcos, rsin, L_TXT, Kb, S_ALL, L_TXT);
  pack_vT_kernel<<<dim3(L_TXT / 64, H_HEADS), b256, 0, stream>>>(EV, VbT, S_ALL, 0);
  pack_vT_kernel<<<dim3(S_IMG / 64, H_HEADS), b256, 0, stream>>>(V0, VbT, S_ALL, L_TXT);
  pack_head_kernel<<<dim3(S_IMG, H_HEADS), dim3(128), 0, stream>>>(Q0, nipq, nullptr, nullptr, 0, IPQ, S_IMG, 0);
  pack_head_kernel<<<dim3(N_IP,  H_HEADS), dim3(128), 0, stream>>>(PK, nipk, nullptr, nullptr, 0, IPK, N_IP, 0);
  pack_vT_kernel<<<dim3(N_IP / 64, H_HEADS), b256, 0, stream>>>(PV, IPVT, N_IP, 0);

  // Attention
  const size_t fa_lds = (size_t)(3 * 128 * 128) * 2;   // 96 KB
  const float scale = 0.08838834764831845f;            // 1/sqrt(128)
  flash_attn_kernel<<<dim3(S_ALL / 128, H_HEADS), b256, fa_lds, stream>>>(Qb, Kb, VbT, O, S_ALL, S_ALL, scale);
  flash_attn_kernel<<<dim3(S_IMG / 128, H_HEADS), b256, fa_lds, stream>>>(IPQ, IPK, IPVT, Oip, S_IMG, N_IP, scale);

  // Merge heads + IP residual
  merge_add_kernel<<<dim3((S_ALL * DMODEL) / 256), b256, 0, stream>>>(O, Oip, Timg, Ttxt);

  // Output projections into d_out (img first, then enc)
  tgemm(Timg, Wo,  bo,  img_out, S_IMG, DMODEL, DMODEL);
  tgemm(Ttxt, Wao, bao, enc_out, L_TXT, DMODEL, DMODEL);
}
